// GIN_73950746902574
// MI455X (gfx1250) — matrix operand order repaired and checked by reading, not compiled, not run on hardware
//
#include <hip/hip_runtime.h>
#include <hip/hip_bf16.h>

typedef __attribute__((ext_vector_type(16))) _Float16 v16h;
typedef __attribute__((ext_vector_type(8)))  _Float16 v8h;
typedef __attribute__((ext_vector_type(4)))  _Float16 v4h;
typedef __attribute__((ext_vector_type(8)))  float    v8f;

#define NNODES 400000
#define NEDGES 800000
#define NGRAPH 16384
#define DH     128     // 2*H
#define NL     4

// ---------------------------------------------------------------------------
// Feature embedding (f16 outputs):
//   h[:,0:64] = atom_emb[x0]; h[:,64:128] = bits(x[1:8]) @ W + b
// ---------------------------------------------------------------------------
__global__ void node_feat_kernel(const int* __restrict__ x,
                                 const float* __restrict__ emb,
                                 const float* __restrict__ lw,
                                 const float* __restrict__ lb,
                                 _Float16* __restrict__ h) {
  long t = (long)blockIdx.x * blockDim.x + threadIdx.x;
  if (t >= (long)NNODES * DH) return;
  int node = (int)(t >> 7);
  int c = (int)(t & 127);
  const int* xr = x + node * 8;
  float out;
  if (c < 64) {
    out = emb[xr[0] * 64 + c];
  } else {
    int co = c - 64;
    float acc = lb[co];
    #pragma unroll
    for (int j = 1; j < 8; ++j) {
      int v = xr[j];
      #pragma unroll
      for (int tb = 0; tb < 8; ++tb) {        // MSB-first (np.unpackbits order)
        float bit = (float)((v >> (7 - tb)) & 1);
        acc = fmaf(bit, lw[((j - 1) * 8 + tb) * 64 + co], acc);
      }
    }
    out = acc;
  }
  h[t] = (_Float16)out;
}

__global__ void edge_feat_kernel(const int* __restrict__ ea,
                                 const float* __restrict__ emb,
                                 const float* __restrict__ lw,
                                 const float* __restrict__ lb,
                                 _Float16* __restrict__ e) {
  long t = (long)blockIdx.x * blockDim.x + threadIdx.x;
  if (t >= (long)NEDGES * DH) return;
  int edge = (int)(t >> 7);
  int c = (int)(t & 127);
  const int* xr = ea + edge * 2;
  float out;
  if (c < 64) {
    out = emb[xr[0] * 64 + c];
  } else {
    int co = c - 64;
    float acc = lb[co];
    int v = xr[1];
    #pragma unroll
    for (int tb = 0; tb < 8; ++tb) {
      float bit = (float)((v >> (7 - tb)) & 1);
      acc = fmaf(bit, lw[tb * 64 + co], acc);
    }
    out = acc;
  }
  e[t] = (_Float16)out;
}

// ---------------------------------------------------------------------------
// agg[i] = (float)h16[i]   (z = (1+eps)*x with eps=0, before message scatter)
// Vectorized: 4 halves in, float4 out per thread.
// ---------------------------------------------------------------------------
__global__ void widen_f16_kernel(const _Float16* __restrict__ s,
                                 float* __restrict__ d, long n4) {
  long t = (long)blockIdx.x * blockDim.x + threadIdx.x;
  if (t >= n4) return;
  v4h v = *(const v4h*)(s + t * 4);
  float4 o;
  o.x = (float)v[0]; o.y = (float)v[1]; o.z = (float)v[2]; o.w = (float)v[3];
  *(float4*)(d + t * 4) = o;
}

// ---------------------------------------------------------------------------
// Message + scatter: agg[dst] += relu(h[src] + e).  One wave per edge, each
// lane handles 4 contiguous channels (f16 gathers, f32 atomic accumulate).
// ---------------------------------------------------------------------------
__global__ void scatter_kernel(const _Float16* __restrict__ h,
                               const _Float16* __restrict__ e,
                               const int* __restrict__ src,
                               const int* __restrict__ dst,
                               float* __restrict__ agg) {
  long t = (long)blockIdx.x * blockDim.x + threadIdx.x;
  if (t >= (long)NEDGES * 32) return;
  int edge = (int)(t >> 5);
  int lane = (int)(t & 31);
  int c = lane * 4;
  int s = src[edge];
  int d = dst[edge];
  v4h hv = *(const v4h*)(h + (long)s * DH + c);
  v4h ev = *(const v4h*)(e + (long)edge * DH + c);
  float m0 = fmaxf((float)hv[0] + (float)ev[0], 0.f);
  float m1 = fmaxf((float)hv[1] + (float)ev[1], 0.f);
  float m2 = fmaxf((float)hv[2] + (float)ev[2], 0.f);
  float m3 = fmaxf((float)hv[3] + (float)ev[3], 0.f);
  float* ap = agg + (long)d * DH + c;
  atomicAdd(ap + 0, m0);
  atomicAdd(ap + 1, m1);
  atomicAdd(ap + 2, m2);
  atomicAdd(ap + 3, m3);
}

// ---------------------------------------------------------------------------
// A-fragment loaders (overloads select f32-with-convert vs direct f16) and
// C stores.  A fragment (16-bit 16x32): halves 0..7 = K kc+off..+7,
// halves 8..15 = K kc+off+16..+23, off = (lane>=16)?8:0.
// ---------------------------------------------------------------------------
union AFrag { v16h v; v8h h8[2]; _Float16 e16[16]; };

__device__ inline void load_afrag(const float* __restrict__ Arow, int kc, int hi,
                                  AFrag& a) {
  const float* p = Arow + kc + hi * 8;
  float4 f0 = *(const float4*)(p + 0);
  float4 f1 = *(const float4*)(p + 4);
  float4 f2 = *(const float4*)(p + 16);
  float4 f3 = *(const float4*)(p + 20);
  a.e16[0]  = (_Float16)f0.x;  a.e16[1]  = (_Float16)f0.y;
  a.e16[2]  = (_Float16)f0.z;  a.e16[3]  = (_Float16)f0.w;
  a.e16[4]  = (_Float16)f1.x;  a.e16[5]  = (_Float16)f1.y;
  a.e16[6]  = (_Float16)f1.z;  a.e16[7]  = (_Float16)f1.w;
  a.e16[8]  = (_Float16)f2.x;  a.e16[9]  = (_Float16)f2.y;
  a.e16[10] = (_Float16)f2.z;  a.e16[11] = (_Float16)f2.w;
  a.e16[12] = (_Float16)f3.x;  a.e16[13] = (_Float16)f3.y;
  a.e16[14] = (_Float16)f3.z;  a.e16[15] = (_Float16)f3.w;
}

__device__ inline void load_afrag(const _Float16* __restrict__ Arow, int kc, int hi,
                                  AFrag& a) {
  const _Float16* p = Arow + kc + hi * 8;     // 16B aligned
  a.h8[0] = *(const v8h*)(p);                 // K = kc+off   .. +7
  a.h8[1] = *(const v8h*)(p + 16);            // K = kc+off+16 .. +23
}

__device__ inline void store_c(float* __restrict__ C, long idx, float v) {
  C[idx] = v;
}
__device__ inline void store_c(_Float16* __restrict__ C, long idx, float v) {
  C[idx] = (_Float16)v;
}

// ---------------------------------------------------------------------------
// WMMA GEMM:  C[M,Nc] = act( (A[M,K] @ W16[K,Nc] + bias) * scale + shift )
// 256 threads = 8 waves; wave w computes a 16x32 output tile (two 16x16
// accumulators sharing one A fragment -> two independent WMMAs per chunk,
// hiding the WMMA hazard window and halving A traffic per FLOP).
// Block tile: 128 rows x 32 cols.  B panel staged in LDS [32][K+8] halves.
// B fragment: lane n=lane&15, same K order as the A fragment.
// Out-of-range rows are clamped (not branched): bogus rows only produce
// bogus D rows, which the store guard drops.
// ---------------------------------------------------------------------------
template <typename TA, typename TC>
__global__ void __launch_bounds__(256) wmma_gemm_kernel(
    const TA* __restrict__ A, const _Float16* __restrict__ W16,
    const float* __restrict__ bias,
    const float* __restrict__ bn_g, const float* __restrict__ bn_b,
    const float* __restrict__ bn_m, const float* __restrict__ bn_v,
    TC* __restrict__ C, int M, int K, int Ncols, int do_relu) {
  __shared__ _Float16 Wlds[32 * (1024 + 8)];   // up to K=1024, padded stride
  const int ldw = K + 8;
  const int ntile = blockIdx.y * 32;

  {  // cooperative fill of the 32-column weight panel, transposed to [n][k]
    int n = threadIdx.x & 31;
    int col = ntile + n;
    bool ok = col < Ncols;
    for (int k = (threadIdx.x >> 5); k < K; k += 8) {
      Wlds[n * ldw + k] = ok ? W16[(long)k * Ncols + col] : (_Float16)0.f;
    }
  }
  __syncthreads();

  const int lane = threadIdx.x & 31;
  const int wave = threadIdx.x >> 5;
  const int hi   = lane >> 4;       // half-wave select
  const int nn   = lane & 15;
  const int mtile = blockIdx.x * 128 + wave * 16;
  int mrow = mtile + nn;
  if (mrow >= M) mrow = M - 1;      // clamp: no branch in the K loop
  const TA* Arow = A + (long)mrow * K;
  const _Float16* Brow0 = &Wlds[nn * ldw];
  const _Float16* Brow1 = &Wlds[(nn + 16) * ldw];

  v8f acc0 = {};
  v8f acc1 = {};
  for (int kc = 0; kc < K; kc += 32) {
    AFrag a;
    load_afrag(Arow, kc, hi, a);
    union { v16h v; v8h h8[2]; } b0, b1;
    const _Float16* bp0 = Brow0 + kc + hi * 8;    // 16B aligned
    const _Float16* bp1 = Brow1 + kc + hi * 8;
    b0.h8[0] = *(const v8h*)(bp0);
    b0.h8[1] = *(const v8h*)(bp0 + 16);
    b1.h8[0] = *(const v8h*)(bp1);
    b1.h8[1] = *(const v8h*)(bp1 + 16);
    acc0 = __builtin_amdgcn_wmma_f32_16x16x32_f16(
        false, a.v, false, b0.v, (short)0, acc0, false, false);
    acc1 = __builtin_amdgcn_wmma_f32_16x16x32_f16(
        false, a.v, false, b1.v, (short)0, acc1, false, false);
  }

  #pragma unroll
  for (int j = 0; j < 2; ++j) {
    int col = ntile + j * 16 + nn;
    bool colok = col < Ncols;
    float bi = (bias != nullptr && colok) ? bias[col] : 0.f;
    float sc = 1.f, sh = 0.f;
    if (bn_g != nullptr && colok) {
      float inv = rsqrtf(bn_v[col] + 1e-5f);
      sc = bn_g[col] * inv;
      sh = bn_b[col] - bn_m[col] * sc;
    }
    #pragma unroll
    for (int r = 0; r < 8; ++r) {
      int mo = mtile + hi * 8 + r;    // C layout: VGPR r -> M = r (+8 for hi)
      if (mo < M && colok) {
        float v = fmaf((j == 0 ? acc0[r] : acc1[r]) + bi, sc, sh);
        if (do_relu) v = fmaxf(v, 0.f);
        store_c(C, (long)mo * Ncols + col, v);
      }
    }
  }
}

// ---------------------------------------------------------------------------
// Mean pool (f16 node features -> f32 atomic accumulate)
// ---------------------------------------------------------------------------
__global__ void pool_accum_kernel(const _Float16* __restrict__ h,
                                  const int* __restrict__ batch,
                                  float* __restrict__ pooled,
                                  float* __restrict__ cnt) {
  long t = (long)blockIdx.x * blockDim.x + threadIdx.x;
  if (t >= (long)NNODES * 32) return;
  int node = (int)(t >> 5);
  int lane = (int)(t & 31);
  int c = lane * 4;
  int b = batch[node];
  v4h v = *(const v4h*)(h + (long)node * DH + c);
  float* pp = pooled + (long)b * DH + c;
  atomicAdd(pp + 0, (float)v[0]);
  atomicAdd(pp + 1, (float)v[1]);
  atomicAdd(pp + 2, (float)v[2]);
  atomicAdd(pp + 3, (float)v[3]);
  if (lane == 0) atomicAdd(cnt + b, 1.f);
}

__global__ void pool_div_kernel(const float* __restrict__ pooled,
                                const float* __restrict__ cnt,
                                float* __restrict__ g) {
  int t = blockIdx.x * blockDim.x + threadIdx.x;
  if (t >= NGRAPH * DH) return;
  g[t] = pooled[t] / fmaxf(cnt[t >> 7], 1.f);
}

// ---------------------------------------------------------------------------
// f32 -> f16 weight conversion
// ---------------------------------------------------------------------------
__global__ void cvt_f16_kernel(const float* __restrict__ s,
                               _Float16* __restrict__ d, int n) {
  int i = blockIdx.x * blockDim.x + threadIdx.x;
  if (i < n) d[i] = (_Float16)s[i];
}

// ---------------------------------------------------------------------------
template <typename TA, typename TC>
static void launch_gemm(const TA* A, const _Float16* W, const float* bias,
                        const float* bg, const float* bb, const float* bm,
                        const float* bv, TC* C, int M, int K, int Nc, int relu,
                        hipStream_t stream) {
  dim3 grid((M + 127) / 128, (Nc + 31) / 32);
  wmma_gemm_kernel<TA, TC><<<grid, 256, 0, stream>>>(A, W, bias, bg, bb, bm, bv,
                                                     C, M, K, Nc, relu);
}

extern "C" void kernel_launch(void* const* d_in, const int* in_sizes, int n_in,
                              void* d_out, int out_size, void* d_ws, size_t ws_size,
                              hipStream_t stream) {
  (void)in_sizes; (void)n_in; (void)out_size; (void)ws_size;

  const int*   x          = (const int*)d_in[0];
  const int*   edge_index = (const int*)d_in[1];
  const int*   edge_attr  = (const int*)d_in[2];
  const int*   batch      = (const int*)d_in[3];
  const float* atom_emb_w = (const float*)d_in[4];
  const float* atom_lin_w = (const float*)d_in[5];
  const float* atom_lin_b = (const float*)d_in[6];
  const float* edge_emb_w = (const float*)d_in[7];
  const float* edge_lin_w = (const float*)d_in[8];
  const float* edge_lin_b = (const float*)d_in[9];
  const float* conv_W1 = (const float*)d_in[10];
  const float* conv_b1 = (const float*)d_in[11];
  const float* conv_g1 = (const float*)d_in[12];
  const float* conv_be1= (const float*)d_in[13];
  const float* conv_m1 = (const float*)d_in[14];
  const float* conv_v1 = (const float*)d_in[15];
  const float* conv_W2 = (const float*)d_in[16];
  const float* conv_b2 = (const float*)d_in[17];
  const float* conv_g2 = (const float*)d_in[18];
  const float* conv_be2= (const float*)d_in[19];
  const float* conv_m2 = (const float*)d_in[20];
  const float* conv_v2 = (const float*)d_in[21];
  const float* lin_W1 = (const float*)d_in[22];
  const float* lin_b1 = (const float*)d_in[23];
  const float* lin_W2 = (const float*)d_in[24];
  const float* lin_b2 = (const float*)d_in[25];
  const float* lin_W3 = (const float*)d_in[26];
  const float* lin_b3 = (const float*)d_in[27];
  const float* lin_W4 = (const float*)d_in[28];
  const float* lin_b4 = (const float*)d_in[29];

  char* wp = (char*)d_ws;
  auto alloc = [&](size_t bytes) -> void* {
    void* p = (void*)wp;
    wp += (bytes + 255) & ~(size_t)255;
    return p;
  };
  _Float16* h16  = (_Float16*)alloc((size_t)NNODES * DH * 2);
  _Float16* e16  = (_Float16*)alloc((size_t)NEDGES * DH * 2);
  _Float16* t16  = (_Float16*)alloc((size_t)NNODES * 256 * 2);
  float*    agg  = (float*)alloc((size_t)NNODES * DH * 4);
  float*  pooled = (float*)alloc((size_t)NGRAPH * DH * 4);
  float*    cnt  = (float*)alloc((size_t)NGRAPH * 4);
  float*    g    = (float*)alloc((size_t)NGRAPH * DH * 4);
  _Float16* cW1h = (_Float16*)alloc((size_t)NL * 128 * 256 * 2);
  _Float16* cW2h = (_Float16*)alloc((size_t)NL * 256 * 128 * 2);
  _Float16* lW1h = (_Float16*)alloc((size_t)128 * 1024 * 2);
  _Float16* lW2h = (_Float16*)alloc((size_t)1024 * 1024 * 2);
  _Float16* lW3h = (_Float16*)alloc((size_t)1024 * 512 * 2);
  _Float16* lW4h = (_Float16*)alloc((size_t)512 * 3 * 2);
  // classifier temps alias buffers that are dead after the conv layers
  _Float16* c1 = t16;               // [B,1024]  (t16 has N*256 halves)
  _Float16* c2 = e16;               // [B,1024]
  _Float16* c3 = (_Float16*)agg;    // [B,512]
  float* out = (float*)d_out;

  auto cvt = [&](const float* s, _Float16* d, int n) {
    cvt_f16_kernel<<<(n + 255) / 256, 256, 0, stream>>>(s, d, n);
  };
  cvt(conv_W1, cW1h, NL * 128 * 256);
  cvt(conv_W2, cW2h, NL * 256 * 128);
  cvt(lin_W1, lW1h, 128 * 1024);
  cvt(lin_W2, lW2h, 1024 * 1024);
  cvt(lin_W3, lW3h, 1024 * 512);
  cvt(lin_W4, lW4h, 512 * 3);

  node_feat_kernel<<<(unsigned)(((long)NNODES * DH + 255) / 256), 256, 0, stream>>>(
      x, atom_emb_w, atom_lin_w, atom_lin_b, h16);
  edge_feat_kernel<<<(unsigned)(((long)NEDGES * DH + 255) / 256), 256, 0, stream>>>(
      edge_attr, edge_emb_w, edge_lin_w, edge_lin_b, e16);

  for (int l = 0; l < NL; ++l) {
    // z = h + sum_j relu(h[src] + e)   (eps = 0)
    long n4 = (long)NNODES * DH / 4;
    widen_f16_kernel<<<(unsigned)((n4 + 255) / 256), 256, 0, stream>>>(h16, agg, n4);
    scatter_kernel<<<(unsigned)(((long)NEDGES * 32 + 255) / 256), 256, 0, stream>>>(
        h16, e16, edge_index, edge_index + NEDGES, agg);
    launch_gemm(agg, cW1h + (size_t)l * 128 * 256, conv_b1 + l * 256,
                conv_g1 + l * 256, conv_be1 + l * 256, conv_m1 + l * 256,
                conv_v1 + l * 256, t16, NNODES, 128, 256, 1, stream);
    launch_gemm(t16, cW2h + (size_t)l * 256 * 128, conv_b2 + l * 128,
                conv_g2 + l * 128, conv_be2 + l * 128, conv_m2 + l * 128,
                conv_v2 + l * 128, h16, NNODES, 256, 128, (l < NL - 1) ? 1 : 0,
                stream);
  }

  hipMemsetAsync(pooled, 0, (size_t)NGRAPH * DH * 4, stream);
  hipMemsetAsync(cnt, 0, (size_t)NGRAPH * 4, stream);
  pool_accum_kernel<<<(unsigned)(((long)NNODES * 32 + 255) / 256), 256, 0, stream>>>(
      h16, batch, pooled, cnt);
  pool_div_kernel<<<(NGRAPH * DH + 255) / 256, 256, 0, stream>>>(pooled, cnt, g);

  launch_gemm(g,  lW1h, lin_b1, (const float*)nullptr, (const float*)nullptr,
              (const float*)nullptr, (const float*)nullptr, c1, NGRAPH, 128, 1024, 1, stream);
  launch_gemm(c1, lW2h, lin_b2, (const float*)nullptr, (const float*)nullptr,
              (const float*)nullptr, (const float*)nullptr, c2, NGRAPH, 1024, 1024, 1, stream);
  launch_gemm(c2, lW3h, lin_b3, (const float*)nullptr, (const float*)nullptr,
              (const float*)nullptr, (const float*)nullptr, c3, NGRAPH, 1024, 512, 1, stream);
  launch_gemm(c3, lW4h, lin_b4, (const float*)nullptr, (const float*)nullptr,
              (const float*)nullptr, (const float*)nullptr, out, NGRAPH, 512, 3, 0, stream);
}
